// AgentEncoderWithRouting_21337397527141
// MI455X (gfx1250) — matrix operand order repaired and checked by reading, not compiled, not run on hardware
//
#include <hip/hip_runtime.h>
#include <hip/hip_bf16.h>
#include <math.h>

// ---- problem constants (match reference) ----
#define B_ 8
#define N_ 4096
#define D_ 768
#define K_ 5
#define ITERS_ 2

typedef __attribute__((ext_vector_type(16))) __bf16 v16bf;
typedef __attribute__((ext_vector_type(8)))  __bf16 v8bf;
typedef __attribute__((ext_vector_type(8)))  float  v8f;

// =====================================================================
// Row LayerNorm (768 wide, 256 threads/block, 3 elems/thread)
// srcRowMod > 0 => source row = blockIdx.x % srcRowMod (broadcast case)
// =====================================================================
template <typename OutT>
__global__ __launch_bounds__(256) void ln_row_kernel(
    const float* __restrict__ in, OutT* __restrict__ out,
    const float* __restrict__ g, const float* __restrict__ bt,
    int srcRowMod)
{
    __shared__ float red[256];
    const int row  = blockIdx.x;
    const int srow = (srcRowMod > 0) ? (row % srcRowMod) : row;
    const float* rp = in + (size_t)srow * D_;
    const int tid = threadIdx.x;

    float x0 = rp[tid], x1 = rp[tid + 256], x2 = rp[tid + 512];
    red[tid] = x0 + x1 + x2;
    __syncthreads();
    for (int o = 128; o > 0; o >>= 1) {
        if (tid < o) red[tid] += red[tid + o];
        __syncthreads();
    }
    const float mean = red[0] * (1.0f / 768.0f);
    __syncthreads();

    const float d0 = x0 - mean, d1 = x1 - mean, d2 = x2 - mean;
    red[tid] = d0 * d0 + d1 * d1 + d2 * d2;
    __syncthreads();
    for (int o = 128; o > 0; o >>= 1) {
        if (tid < o) red[tid] += red[tid + o];
        __syncthreads();
    }
    const float rstd = rsqrtf(red[0] * (1.0f / 768.0f) + 1e-5f);

    OutT* op = out + (size_t)row * D_;
    op[tid]       = (OutT)(d0 * rstd * g[tid]       + bt[tid]);
    op[tid + 256] = (OutT)(d1 * rstd * g[tid + 256] + bt[tid + 256]);
    op[tid + 512] = (OutT)(d2 * rstd * g[tid + 512] + bt[tid + 512]);
}

// =====================================================================
// Cast Wk, Wv (f32 row-major [D,D]) to bf16
// =====================================================================
__global__ __launch_bounds__(256) void cast2_bf16(
    const float* __restrict__ a, const float* __restrict__ b,
    __bf16* __restrict__ oa, __bf16* __restrict__ ob, int n)
{
    int t = blockIdx.x * 256 + threadIdx.x;
    if (t < n) { oa[t] = (__bf16)a[t]; ob[t] = (__bf16)b[t]; }
}

// =====================================================================
// Fused k/v projection: C_w = Xn @ W_w^T, w in {k,v}
//   Xn: [B*N, D] bf16 row-major   W: [D,D] bf16 row-major (row e = B-col e)
// Block tile 128(M) x 64(N), K-step 32, 8 waves: 4(M)x2(N) each 32x32.
// Both weights share one A-tile load (halves the dominant HBM stream).
// =====================================================================
__global__ __launch_bounds__(256) void gemm_kv_wmma(
    const __bf16* __restrict__ xn,
    const __bf16* __restrict__ wk, const __bf16* __restrict__ wv,
    __bf16* __restrict__ kout, __bf16* __restrict__ vout)
{
    __shared__ __bf16 As[128][40];      // 32 K + 8 pad halfs per row
    __shared__ __bf16 Bs[2][64][40];    // column-major tiles per weight

    const int tid  = threadIdx.x;
    const int lane = tid & 31;
    const int wave = tid >> 5;
    const int hl   = lane >> 4;         // lane half (0: lanes 0-15, 1: 16-31)
    const int ll   = lane & 15;
    const int m0   = blockIdx.y * 128;
    const int n0   = blockIdx.x * 64;
    const int wm   = (wave & 3) * 32;
    const int wn   = (wave >> 2) * 32;

    const __bf16* wsel[2] = { wk, wv };

    v8f acc[2][2][2];
#pragma unroll
    for (int w = 0; w < 2; ++w)
#pragma unroll
        for (int i = 0; i < 2; ++i)
#pragma unroll
            for (int j = 0; j < 2; ++j)
#pragma unroll
                for (int r = 0; r < 8; ++r) acc[w][i][j][r] = 0.0f;

    for (int k0 = 0; k0 < D_; k0 += 32) {
        __syncthreads();   // previous-iteration fragment reads done
        // --- A tile: 128 rows x 32 halfs = 512 x 16B chunks ---
#pragma unroll
        for (int c = tid; c < 512; c += 256) {
            const int r = c >> 2, s = c & 3;
            const uint4 u = *(const uint4*)(xn + (size_t)(m0 + r) * D_ + k0 + s * 8);
            *(uint4*)(&As[r][s * 8]) = u;
        }
        // --- B tiles: 2 weights x 64 cols x 32 halfs = 512 chunks ---
#pragma unroll
        for (int c = tid; c < 512; c += 256) {
            const int w   = c >> 8;
            const int col = (c >> 2) & 63;
            const int s   = c & 3;
            const uint4 u = *(const uint4*)(wsel[w] + (size_t)(n0 + col) * D_ + k0 + s * 8);
            *(uint4*)(&Bs[w][col][s * 8]) = u;
        }
        __syncthreads();

        // A fragments (ISA 7.12.2: lanes 0-15 K=0-7/16-23, lanes 16-31 K=8-15/24-31)
        v16bf afr[2];
#pragma unroll
        for (int i = 0; i < 2; ++i) {
            const int row = wm + i * 16 + ll;
            const int ka  = hl ? 8 : 0, kb = hl ? 24 : 16;
            v8bf a0 = *(const v8bf*)(&As[row][ka]);
            v8bf a1 = *(const v8bf*)(&As[row][kb]);
#pragma unroll
            for (int e = 0; e < 8; ++e) { afr[i][e] = a0[e]; afr[i][e + 8] = a1[e]; }
        }
        // B fragments (lanes 0-15: K=0-7/16-23 of col ll; lanes 16-31: K=8-15/24-31)
        v16bf bfr[2][2];
#pragma unroll
        for (int w = 0; w < 2; ++w)
#pragma unroll
            for (int j = 0; j < 2; ++j) {
                const int col = wn + j * 16 + ll;
                const int kb0 = hl ? 8 : 0;
                v8bf b0 = *(const v8bf*)(&Bs[w][col][kb0]);
                v8bf b1 = *(const v8bf*)(&Bs[w][col][kb0 + 16]);
#pragma unroll
                for (int e = 0; e < 8; ++e) { bfr[w][j][e] = b0[e]; bfr[w][j][e + 8] = b1[e]; }
            }
#pragma unroll
        for (int w = 0; w < 2; ++w)
#pragma unroll
            for (int i = 0; i < 2; ++i)
#pragma unroll
                for (int j = 0; j < 2; ++j)
                    acc[w][i][j] = __builtin_amdgcn_wmma_f32_16x16x32_bf16(
                        false, afr[i], false, bfr[w][j],
                        (short)0, acc[w][i][j], false, false);
    }

    __bf16* outp[2] = { kout, vout };
#pragma unroll
    for (int w = 0; w < 2; ++w)
#pragma unroll
        for (int i = 0; i < 2; ++i)
#pragma unroll
            for (int j = 0; j < 2; ++j)
#pragma unroll
                for (int r = 0; r < 8; ++r) {
                    const int row = m0 + wm + i * 16 + r + (hl ? 8 : 0);
                    const int col = n0 + wn + j * 16 + ll;
                    outp[w][(size_t)row * D_ + col] = (__bf16)acc[w][i][j][r];
                }
}

// =====================================================================
// logits[b,j,n] = scale * sum_d q[b,j,d] * k[b,n,d]   (K=5 padded to 16)
// WMMA: A = k rows (16 tokens / wave), B = q^T staged zero-padded in LDS.
// grid (N/128, B), 256 threads (8 waves x one 16-row tile)
// =====================================================================
__global__ __launch_bounds__(256) void logits_wmma(
    const __bf16* __restrict__ kb, const float* __restrict__ q,
    float* __restrict__ logits, float scale)
{
    __shared__ __bf16 Qs[16][D_];
    const int b   = blockIdx.y;
    const int tid = threadIdx.x;
    for (int i = tid; i < 16 * D_; i += 256) {
        const int j = i / D_, d = i % D_;
        Qs[j][d] = (j < K_) ? (__bf16)q[((size_t)b * K_ + j) * D_ + d] : (__bf16)0.0f;
    }
    __syncthreads();

    const int lane = tid & 31, wave = tid >> 5;
    const int hl = lane >> 4, ll = lane & 15;
    const int nbase = blockIdx.x * 128 + wave * 16;
    const __bf16* arow = kb + ((size_t)b * N_ + nbase + ll) * D_;

    v8f acc;
#pragma unroll
    for (int r = 0; r < 8; ++r) acc[r] = 0.0f;

    for (int k0 = 0; k0 < D_; k0 += 32) {
        const int ka = k0 + (hl ? 8 : 0), ka2 = k0 + (hl ? 24 : 16);
        v8bf a0 = *(const v8bf*)(arow + ka);
        v8bf a1 = *(const v8bf*)(arow + ka2);
        v16bf afr;
#pragma unroll
        for (int e = 0; e < 8; ++e) { afr[e] = a0[e]; afr[e + 8] = a1[e]; }
        const int kb0 = k0 + (hl ? 8 : 0);
        v8bf b0 = *(const v8bf*)(&Qs[ll][kb0]);
        v8bf b1 = *(const v8bf*)(&Qs[ll][kb0 + 16]);
        v16bf bfr;
#pragma unroll
        for (int e = 0; e < 8; ++e) { bfr[e] = b0[e]; bfr[e + 8] = b1[e]; }
        acc = __builtin_amdgcn_wmma_f32_16x16x32_bf16(
            false, afr, false, bfr, (short)0, acc, false, false);
    }

    if (ll < K_) {
#pragma unroll
        for (int r = 0; r < 8; ++r) {
            const int n = nbase + r + (hl ? 8 : 0);
            logits[((size_t)b * K_ + ll) * N_ + n] = acc[r] * scale;
        }
    }
}

// =====================================================================
// softmax over slot axis (K=5) per (b,n)
// =====================================================================
__global__ __launch_bounds__(256) void softmax_k(
    const float* __restrict__ logits, float* __restrict__ attn)
{
    const int t = blockIdx.x * 256 + threadIdx.x;
    if (t >= B_ * N_) return;
    const int b = t / N_, n = t % N_;
    const float* lp = logits + (size_t)b * K_ * N_ + n;
    float v[K_], mx = -1e30f;
#pragma unroll
    for (int j = 0; j < K_; ++j) { v[j] = lp[(size_t)j * N_]; mx = fmaxf(mx, v[j]); }
    float s = 0.0f;
#pragma unroll
    for (int j = 0; j < K_; ++j) { v[j] = __expf(v[j] - mx); s += v[j]; }
    const float inv = 1.0f / s;
    float* ap = attn + (size_t)b * K_ * N_ + n;
#pragma unroll
    for (int j = 0; j < K_; ++j) ap[(size_t)j * N_] = v[j] * inv;
}

// =====================================================================
// updates[b,j,d] = sum_n attn[b,j,n] * v[b,n,d]  (M=5 -> bandwidth bound:
// stream v coalesced, attn tiles staged in LDS). grid (D/256, B)
// =====================================================================
__global__ __launch_bounds__(256) void updates_valu(
    const float* __restrict__ attn, const __bf16* __restrict__ vb,
    float* __restrict__ upd)
{
    __shared__ float sa[K_][256];
    const int b = blockIdx.y;
    const int d = blockIdx.x * 256 + threadIdx.x;
    const float*  ab   = attn + (size_t)b * K_ * N_;
    const __bf16* vcol = vb + (size_t)b * N_ * D_ + d;

    float acc[K_];
#pragma unroll
    for (int j = 0; j < K_; ++j) acc[j] = 0.0f;

    for (int nt = 0; nt < N_; nt += 256) {
#pragma unroll
        for (int j = 0; j < K_; ++j)
            sa[j][threadIdx.x] = ab[(size_t)j * N_ + nt + threadIdx.x];
        __syncthreads();
        __builtin_prefetch(vcol + (size_t)(nt + 256) * D_, 0, 0); // global_prefetch_b8
        for (int n = 0; n < 256; ++n) {
            const float vv = (float)vcol[(size_t)(nt + n) * D_];
#pragma unroll
            for (int j = 0; j < K_; ++j) acc[j] += sa[j][n] * vv;
        }
        __syncthreads();
    }
#pragma unroll
    for (int j = 0; j < K_; ++j)
        upd[((size_t)b * K_ + j) * D_ + d] = acc[j];
}

// =====================================================================
// Generic small GEMM (M<=48): C = act(A @ W^T + bias)
//   A: [M,Kc] f32, W: [Ncols,Kc] f32, one wave per 16x16 tile.
// f32 -> bf16 fragment conversion on the fly; WMMA accumulation in f32.
// =====================================================================
__global__ __launch_bounds__(32) void gemm_small_wmma(
    const float* __restrict__ A, const float* __restrict__ W,
    const float* __restrict__ bias, float* __restrict__ C,
    int M, int Ncols, int Kc, int act)
{
    const int lane = threadIdx.x & 31;
    const int hl = lane >> 4, ll = lane & 15;
    const int m0 = blockIdx.y * 16, n0 = blockIdx.x * 16;
    const int row = m0 + ll;  const bool rok = row < M;
    const int col = n0 + ll;  const bool cok = col < Ncols;
    const float* ap = A + (size_t)(rok ? row : 0) * Kc;
    const float* wp = W + (size_t)(cok ? col : 0) * Kc;

    v8f acc;
#pragma unroll
    for (int r = 0; r < 8; ++r) acc[r] = 0.0f;

    for (int k0 = 0; k0 < Kc; k0 += 32) {
        const int ka = k0 + (hl ? 8 : 0), ka2 = k0 + (hl ? 24 : 16);
        v16bf afr, bfr;
#pragma unroll
        for (int e = 0; e < 8; ++e) {
            afr[e]     = rok ? (__bf16)ap[ka + e]  : (__bf16)0.0f;
            afr[e + 8] = rok ? (__bf16)ap[ka2 + e] : (__bf16)0.0f;
        }
        const int kb0 = k0 + (hl ? 8 : 0);
#pragma unroll
        for (int e = 0; e < 16; ++e)
            bfr[e] = cok ? (__bf16)wp[kb0 + (e & 7) + ((e >> 3) << 4)] : (__bf16)0.0f;
        acc = __builtin_amdgcn_wmma_f32_16x16x32_bf16(
            false, afr, false, bfr, (short)0, acc, false, false);
    }

    if (cok) {
        const float bv = bias ? bias[col] : 0.0f;
#pragma unroll
        for (int r = 0; r < 8; ++r) {
            const int rr = m0 + r + hl * 8;
            if (rr < M) {
                float v = acc[r] + bv;
                if (act == 1) v = v > 0.0f ? v : 0.0f;
                C[(size_t)rr * Ncols + col] = v;
            }
        }
    }
}

// =====================================================================
// GRU gate fusion: h = (1-z)*n + z*h_prev
// =====================================================================
__global__ __launch_bounds__(256) void gru_combine(
    const float* __restrict__ gi, const float* __restrict__ gh,
    const float* __restrict__ hprev, float* __restrict__ h)
{
    const int t = blockIdx.x * 256 + threadIdx.x;
    if (t >= B_ * K_ * D_) return;
    const int row = t / D_, d = t % D_;
    const float* gir = gi + (size_t)row * (3 * D_);
    const float* ghr = gh + (size_t)row * (3 * D_);
    const float r = 1.0f / (1.0f + __expf(-(gir[d] + ghr[d])));
    const float z = 1.0f / (1.0f + __expf(-(gir[D_ + d] + ghr[D_ + d])));
    const float nn = tanhf(gir[2 * D_ + d] + r * ghr[2 * D_ + d]);
    h[t] = (1.0f - z) * nn + z * hprev[t];
}

// =====================================================================
// slots = h + y
// =====================================================================
__global__ __launch_bounds__(256) void add_kernel(
    const float* __restrict__ a, const float* __restrict__ b,
    float* __restrict__ o, int n)
{
    const int t = blockIdx.x * 256 + threadIdx.x;
    if (t < n) o[t] = a[t] + b[t];
}

// =====================================================================
// Pack outputs: [slots (B*K*D)] ++ [attn_weights (B,N,K) = attn^T]
// =====================================================================
__global__ __launch_bounds__(256) void write_out(
    const float* __restrict__ slots, const float* __restrict__ attn,
    float* __restrict__ out)
{
    const int t = blockIdx.x * 256 + threadIdx.x;
    const int nslots = B_ * K_ * D_;
    const int total  = nslots + B_ * N_ * K_;
    if (t >= total) return;
    if (t < nslots) { out[t] = slots[t]; return; }
    const int i = t - nslots;
    const int j = i % K_;
    const int bn = i / K_;
    const int b = bn / N_, n = bn % N_;
    out[t] = attn[((size_t)b * K_ + j) * N_ + n];
}

// =====================================================================
// Host launch
// =====================================================================
static inline size_t align256(size_t x) { return (x + 255) & ~(size_t)255; }

extern "C" void kernel_launch(void* const* d_in, const int* in_sizes, int n_in,
                              void* d_out, int out_size, void* d_ws, size_t ws_size,
                              hipStream_t stream) {
    (void)in_sizes; (void)n_in; (void)out_size; (void)ws_size;
    const float* x          = (const float*)d_in[0];
    const float* slot_inits = (const float*)d_in[1];
    const float* Wq         = (const float*)d_in[2];
    const float* Wk         = (const float*)d_in[3];
    const float* Wv         = (const float*)d_in[4];
    const float* W_ih       = (const float*)d_in[5];
    const float* W_hh       = (const float*)d_in[6];
    const float* b_ih       = (const float*)d_in[7];
    const float* b_hh       = (const float*)d_in[8];
    const float* mlp_ln_g   = (const float*)d_in[9];
    const float* mlp_ln_b   = (const float*)d_in[10];
    const float* W1         = (const float*)d_in[11];
    const float* b1         = (const float*)d_in[12];
    const float* W2         = (const float*)d_in[13];
    const float* b2         = (const float*)d_in[14];
    const float* in_ln_g    = (const float*)d_in[15];
    const float* in_ln_b    = (const float*)d_in[16];
    const float* slot_ln_g  = (const float*)d_in[17];
    const float* slot_ln_b  = (const float*)d_in[18];
    float* out = (float*)d_out;

    // ---- workspace carve-up ----
    char* ws = (char*)d_ws;
    size_t off = 0;
    auto take = [&](size_t bytes) { char* p = ws + off; off += align256(bytes); return p; };
    __bf16* xn   = (__bf16*)take((size_t)B_ * N_ * D_ * 2);
    __bf16* kbuf = (__bf16*)take((size_t)B_ * N_ * D_ * 2);
    __bf16* vbuf = (__bf16*)take((size_t)B_ * N_ * D_ * 2);
    __bf16* wkb  = (__bf16*)take((size_t)D_ * D_ * 2);
    __bf16* wvb  = (__bf16*)take((size_t)D_ * D_ * 2);
    float* slots = (float*)take((size_t)B_ * K_ * D_ * 4);
    float* qbuf  = (float*)take((size_t)B_ * K_ * D_ * 4);
    float* logit = (float*)take((size_t)B_ * K_ * N_ * 4);
    float* attn  = (float*)take((size_t)B_ * K_ * N_ * 4);
    float* upd   = (float*)take((size_t)B_ * K_ * D_ * 4);
    float* gi    = (float*)take((size_t)B_ * K_ * 3 * D_ * 4);
    float* gh    = (float*)take((size_t)B_ * K_ * 3 * D_ * 4);
    float* hbuf  = (float*)take((size_t)B_ * K_ * D_ * 4);
    float* yln   = (float*)take((size_t)B_ * K_ * D_ * 4);
    float* hid   = (float*)take((size_t)B_ * K_ * 2 * D_ * 4);
    float* ybuf  = (float*)take((size_t)B_ * K_ * D_ * 4);

    const float scale = 1.0f / sqrtf((float)D_);

    // 1) x_n = LN(x) -> bf16
    ln_row_kernel<__bf16><<<B_ * N_, 256, 0, stream>>>(x, xn, in_ln_g, in_ln_b, 0);
    // 2) slots = LN(broadcast slot_inits)
    ln_row_kernel<float><<<B_ * K_, 256, 0, stream>>>(slot_inits, slots, slot_ln_g, slot_ln_b, K_);
    // 3) weights -> bf16
    cast2_bf16<<<(D_ * D_ + 255) / 256, 256, 0, stream>>>(Wk, Wv, wkb, wvb, D_ * D_);
    // 4) fused k/v projection (WMMA, dominant GEMMs)
    gemm_kv_wmma<<<dim3(D_ / 64, (B_ * N_) / 128), 256, 0, stream>>>(xn, wkb, wvb, kbuf, vbuf);

    const int Mrows = B_ * K_; // 40
    for (int it = 0; it < ITERS_; ++it) {
        // q = slots @ Wq^T
        gemm_small_wmma<<<dim3(D_ / 16, (Mrows + 15) / 16), 32, 0, stream>>>(
            slots, Wq, nullptr, qbuf, Mrows, D_, D_, 0);
        // logits = scale * q @ k^T (softmax over K axis)
        logits_wmma<<<dim3(N_ / 128, B_), 256, 0, stream>>>(kbuf, qbuf, logit, scale);
        softmax_k<<<(B_ * N_ + 255) / 256, 256, 0, stream>>>(logit, attn);
        // updates = attn @ v
        updates_valu<<<dim3(D_ / 256, B_), 256, 0, stream>>>(attn, vbuf, upd);
        // GRU
        gemm_small_wmma<<<dim3((3 * D_) / 16, (Mrows + 15) / 16), 32, 0, stream>>>(
            upd, W_ih, b_ih, gi, Mrows, 3 * D_, D_, 0);
        gemm_small_wmma<<<dim3((3 * D_) / 16, (Mrows + 15) / 16), 32, 0, stream>>>(
            slots, W_hh, b_hh, gh, Mrows, 3 * D_, D_, 0);
        gru_combine<<<(Mrows * D_ + 255) / 256, 256, 0, stream>>>(gi, gh, slots, hbuf);
        // MLP with residual
        ln_row_kernel<float><<<Mrows, 256, 0, stream>>>(hbuf, yln, mlp_ln_g, mlp_ln_b, 0);
        gemm_small_wmma<<<dim3((2 * D_) / 16, (Mrows + 15) / 16), 32, 0, stream>>>(
            yln, W1, b1, hid, Mrows, 2 * D_, D_, 1);
        gemm_small_wmma<<<dim3(D_ / 16, (Mrows + 15) / 16), 32, 0, stream>>>(
            hid, W2, b2, ybuf, Mrows, D_, 2 * D_, 0);
        add_kernel<<<(Mrows * D_ + 255) / 256, 256, 0, stream>>>(hbuf, ybuf, slots, Mrows * D_);
    }

    const int total = B_ * K_ * D_ + B_ * N_ * K_;
    write_out<<<(total + 255) / 256, 256, 0, stream>>>(slots, attn, out);
}
